// computeRNN_84344567759625
// MI455X (gfx1250) — compile-verified, matrix operand order checked by reading, not run on hardware
//
#include <hip/hip_runtime.h>
#include <hip/hip_bf16.h>
#include <math.h>

// CDNA5 / gfx1250. wave32. BF16 WMMA 16x16x32, FP32 accumulate.

typedef __attribute__((ext_vector_type(16))) __bf16 v16bf;
typedef __attribute__((ext_vector_type(8)))  __bf16 v8bf;
typedef __attribute__((ext_vector_type(4)))  __bf16 v4bf;
typedef __attribute__((ext_vector_type(8)))  float  v8f;

namespace {
constexpr int T_STEPS = 512;
constexpr int BATCH   = 256;
constexpr int IN_F    = 1024;
constexpr int HID     = 1024;
constexpr int NCLS    = 512;
constexpr int KCAT    = IN_F + HID;   // 2048, matches W_ih layout [HID, IN_F+HID]
constexpr int RS      = 1032;         // padded LDS row stride (halfwords): bank-conflict-free b128
constexpr int OS_S    = 516;          // padded o-row stride (floats); 16*OS_S*4 == 16*RS*2
}

// ---------------------------------------------------------------------------
// Kernel 0: one-time FP32 -> BF16 weight conversion into workspace.
// ---------------------------------------------------------------------------
__global__ void rnn_wcvt_kernel(const float* __restrict__ W_ih,
                                const float* __restrict__ W_ho,
                                __bf16* __restrict__ wih_bf,
                                __bf16* __restrict__ who_bf) {
  const int nIh = HID * KCAT;     // 2,097,152
  const int nHo = NCLS * HID;     //   524,288
  for (int i = blockIdx.x * blockDim.x + threadIdx.x; i < nIh + nHo;
       i += gridDim.x * blockDim.x) {
    if (i < nIh) wih_bf[i] = (__bf16)W_ih[i];
    else         who_bf[i - nIh] = (__bf16)W_ho[i - nIh];
  }
}

// ---------------------------------------------------------------------------
// WMMA tile helpers.
// A 16x32 bf16 tile per lane: two 16B runs at halfword offsets {0,+16} from
// rowBase(m16) + 32*kk + kb, where kb = 8*(lane>=16).  B (32x16, = W rows) uses
// the identical per-lane pattern with n playing the role of m.
// ---------------------------------------------------------------------------
__device__ __forceinline__ v16bf ld_tile16(const __bf16* p) {
  v8bf lo = *(const v8bf*)p;
  v8bf hi = *(const v8bf*)(p + 16);
  return __builtin_shufflevector(lo, hi, 0,1,2,3,4,5,6,7,8,9,10,11,12,13,14,15);
}

__device__ __forceinline__ v8f wmma_bf16(v16bf a, v16bf b, v8f c) {
  return __builtin_amdgcn_wmma_f32_16x16x32_bf16(false, a, false, b,
                                                 (short)0, c, false, false);
}

// Branch-free tanh from hardware transcendentals:
//   tanh(x) = 1 - 2/(exp(2x)+1).
// v_exp_f32 + v_rcp_f32; saturates correctly (exp->inf => 1, exp->0 => -1).
// Result is rounded to bf16 downstream, so ~2-3 ulp fp32 error is irrelevant.
__device__ __forceinline__ float fast_tanh(float xv) {
  float e = __expf(2.0f * xv);
  return 1.0f - 2.0f * __builtin_amdgcn_rcpf(e + 1.0f);
}

// ---------------------------------------------------------------------------
// Persistent RNN kernel: 16 blocks x 1024 threads (32 waves / WGP).
// Block b owns batch rows [16b, 16b+16); the whole T loop runs in-kernel.
// h lives in LDS (bf16) for the entire run; no cross-block dependencies.
// ---------------------------------------------------------------------------
__global__ void __launch_bounds__(1024)
rnn_persistent_kernel(const float*  __restrict__ x,      // [T,B,IN_F]
                      const float*  __restrict__ h0,     // [B,HID]
                      const float*  __restrict__ b_ih,   // [HID]
                      const float*  __restrict__ b_ho,   // [NCLS]
                      const __bf16* __restrict__ Wih,    // [HID,KCAT] bf16
                      const __bf16* __restrict__ Who,    // [NCLS,HID] bf16
                      float* __restrict__ y_out,         // [T,B,NCLS]
                      float* __restrict__ h_out)         // [B,HID]
{
  __shared__ __bf16 hb[16 * RS];                                   // 33,024 B
  __shared__ __align__(16) unsigned char xo[16 * RS * 2];          // 33,024 B
  __bf16* xs = (__bf16*)xo;        // staged x_t   (bf16), stride RS
  float*  os = (float*)xo;         // o logits     (f32), stride OS_S (aliased)

  const int tid   = threadIdx.x;
  const int lane  = tid & 31;
  const int wv    = tid >> 5;            // 0..31
  const int m16   = lane & 15;           // A-row / B-row / D-col index
  const int kb    = (lane >> 4) << 3;    // K sub-base: 0 or 8 halfwords
  const int hi8   = (lane >> 4) << 3;    // D-row offset: 0 or 8
  const int brow0 = blockIdx.x * 16;

  // ---- stage h0 (fp32 -> bf16) into LDS --------------------------------
  for (int j = 0; j < 16; ++j)
    hb[j * RS + tid] = (__bf16)h0[(size_t)(brow0 + j) * HID + tid];

  // ---- per-wave constants ----------------------------------------------
  const int nt0 = wv * 2, nt1 = nt0 + 1;          // phase-1 N tiles (64 total)
  const float bias0 = b_ih[nt0 * 16 + m16];
  const float bias1 = b_ih[nt1 * 16 + m16];
  const float biasO = b_ho[wv * 16 + m16];        // phase-2 N tile (32 total)

  const __bf16* wihRow0 = Wih + (size_t)(nt0 * 16 + m16) * KCAT + kb;
  const __bf16* wihRow1 = Wih + (size_t)(nt1 * 16 + m16) * KCAT + kb;
  const __bf16* whoRow  = Who + (size_t)(wv  * 16 + m16) * HID  + kb;
  const int aOff = m16 * RS + kb;                 // per-lane A row base (LDS)

  for (int t = 0; t < T_STEPS; ++t) {
    // ---- stage x_t -> LDS (bf16): float4 loads, b64 packed stores -------
    {
      const float4* xrow4 =
          (const float4*)(x + ((size_t)t * BATCH + brow0) * IN_F);
      #pragma unroll
      for (int k = 0; k < 4; ++k) {
        const int i  = k * 1024 + tid;   // 0..4095 float4s over 16x1024
        const int r  = i >> 8;           // row 0..15 (256 float4 per row)
        const int c4 = i & 255;
        const float4 f = xrow4[i];
        v4bf b;
        b[0] = (__bf16)f.x; b[1] = (__bf16)f.y;
        b[2] = (__bf16)f.z; b[3] = (__bf16)f.w;
        *(v4bf*)(xs + r * RS + c4 * 4) = b;
      }
    }
    __syncthreads();                                              // (A)

    // ---- phase 1: a = [x|h] @ Wih^T + b_ih ; K = 2048 -------------------
    v8f acc0 = {bias0, bias0, bias0, bias0, bias0, bias0, bias0, bias0};
    v8f acc1 = {bias1, bias1, bias1, bias1, bias1, bias1, bias1, bias1};
    #pragma unroll 2
    for (int kk = 0; kk < 32; ++kk) {                 // x half of K
      v16bf A  = ld_tile16(xs + aOff + kk * 32);
      v16bf B0 = ld_tile16(wihRow0 + kk * 32);
      v16bf B1 = ld_tile16(wihRow1 + kk * 32);
      acc0 = wmma_bf16(A, B0, acc0);
      acc1 = wmma_bf16(A, B1, acc1);
    }
    #pragma unroll 2
    for (int kk = 0; kk < 32; ++kk) {                 // h half of K
      v16bf A  = ld_tile16(hb + aOff + kk * 32);
      v16bf B0 = ld_tile16(wihRow0 + IN_F + kk * 32);
      v16bf B1 = ld_tile16(wihRow1 + IN_F + kk * 32);
      acc0 = wmma_bf16(A, B0, acc0);
      acc1 = wmma_bf16(A, B1, acc1);
    }
    float ht0[8], ht1[8];
    #pragma unroll
    for (int r = 0; r < 8; ++r) {
      ht0[r] = fast_tanh(acc0[r]);
      ht1[r] = fast_tanh(acc1[r]);
    }
    __syncthreads();                                              // (B) hb reads done

    // ---- write h_new back to LDS (D layout: n=lane&15, m=r+hi8) ---------
    #pragma unroll
    for (int r = 0; r < 8; ++r) {
      const int mr = r + hi8;
      hb[mr * RS + nt0 * 16 + m16] = (__bf16)ht0[r];
      hb[mr * RS + nt1 * 16 + m16] = (__bf16)ht1[r];
    }
    if (t == T_STEPS - 1) {
      #pragma unroll
      for (int r = 0; r < 8; ++r) {
        const int mr = r + hi8;
        h_out[(size_t)(brow0 + mr) * HID + nt0 * 16 + m16] = ht0[r];
        h_out[(size_t)(brow0 + mr) * HID + nt1 * 16 + m16] = ht1[r];
      }
    }
    __syncthreads();                                              // (C)

    // ---- phase 2: o = h_new @ Who^T + b_ho ; K = 1024 -------------------
    v8f accO = {biasO, biasO, biasO, biasO, biasO, biasO, biasO, biasO};
    #pragma unroll 2
    for (int kk = 0; kk < 32; ++kk) {
      v16bf A = ld_tile16(hb + aOff + kk * 32);
      v16bf B = ld_tile16(whoRow + kk * 32);
      accO = wmma_bf16(A, B, accO);
    }
    {
      const int ncol = wv * 16 + m16;
      #pragma unroll
      for (int r = 0; r < 8; ++r) os[(r + hi8) * OS_S + ncol] = accO[r];
    }

    // ---- prefetch next timestep's x slice (global_prefetch_b8) ----------
    if (t + 1 < T_STEPS) {
      const char* nx =
          (const char*)(x + ((size_t)(t + 1) * BATCH + brow0) * IN_F);
      __builtin_prefetch(nx + (size_t)tid * 64, 0, 3);  // 1024 x 64B = 64 KB
    }
    __syncthreads();                                              // (D)

    // ---- softmax (waves 0..15, one row each; wave32 shfl reductions) ----
    if (wv < 16) {
      const float* orow = os + wv * OS_S;
      float v[16];
      float mx = -3.0e38f;
      #pragma unroll
      for (int j = 0; j < 16; ++j) { v[j] = orow[lane + 32 * j]; mx = fmaxf(mx, v[j]); }
      #pragma unroll
      for (int off = 16; off; off >>= 1) mx = fmaxf(mx, __shfl_xor(mx, off, 32));
      float s = 0.0f;
      #pragma unroll
      for (int j = 0; j < 16; ++j) { v[j] = __expf(v[j] - mx); s += v[j]; }
      #pragma unroll
      for (int off = 16; off; off >>= 1) s += __shfl_xor(s, off, 32);
      const float inv = 1.0f / s;
      float* yrow = y_out + ((size_t)t * BATCH + brow0 + wv) * NCLS;
      #pragma unroll
      for (int j = 0; j < 16; ++j) yrow[lane + 32 * j] = v[j] * inv;
    }
    __syncthreads();                                              // (E) xs/os alias
  }
}

// ---------------------------------------------------------------------------
extern "C" void kernel_launch(void* const* d_in, const int* in_sizes, int n_in,
                              void* d_out, int out_size, void* d_ws, size_t ws_size,
                              hipStream_t stream) {
  const float* input     = (const float*)d_in[0];   // [T,B,IN_F]
  const float* pre_state = (const float*)d_in[1];   // [B,HID]
  const float* W_ih      = (const float*)d_in[2];   // [HID,KCAT]
  const float* b_ih      = (const float*)d_in[3];   // [HID]
  const float* W_ho      = (const float*)d_in[4];   // [NCLS,HID]
  const float* b_ho      = (const float*)d_in[5];   // [NCLS]

  float* out   = (float*)d_out;
  float* y_out = out;                                        // [T,B,NCLS]
  float* h_out = out + (size_t)T_STEPS * BATCH * NCLS;       // [B,HID]

  __bf16* wih_bf = (__bf16*)d_ws;                            // 4 MB
  __bf16* who_bf = wih_bf + (size_t)HID * KCAT;              // 1 MB

  rnn_wcvt_kernel<<<dim3(4096), dim3(256), 0, stream>>>(W_ih, W_ho, wih_bf, who_bf);
  rnn_persistent_kernel<<<dim3(16), dim3(1024), 0, stream>>>(
      input, pre_state, b_ih, b_ho, wih_bf, who_bf, y_out, h_out);
}
